// AwkwardRNNDoubleJagged_438086664429
// MI455X (gfx1250) — 8-bit matrix operand order repaired and checked by reading, compiled, not run on hardware
//
#include <hip/hip_runtime.h>
#include <math.h>


#define NWG   16
#define TPB   256
#define PNUM  256
#define FLEN  128
#define HS2   512
#define OUTN  10

#define SLAB_BYTES   65536
#define CHUNK_BYTES  (3*SLAB_BYTES)
#define WS_SLABS     (NWG*CHUNK_BYTES)
#define SOFF         WS_SLABS
#define ST_WORDS     2306

typedef __attribute__((ext_vector_type(8))) int   v8i;
typedef __attribute__((ext_vector_type(8))) float v8f;

__device__ __forceinline__ unsigned char f32_to_e4m3(float f) {
  unsigned u  = __float_as_uint(f);
  unsigned s  = (u >> 24) & 0x80u;
  unsigned ae = (u >> 23) & 0xFFu;
  unsigned man = u & 0x7FFFFFu;
  if (ae == 0xFFu) return (unsigned char)(s | 0x7E);
  if (ae == 0u)    return (unsigned char)s;
  int e8 = (int)ae - 127 + 7;
  if (e8 >= 16) return (unsigned char)(s | 0x7E);
  unsigned m = 0x800000u | man;
  unsigned rshift = 20u;
  if (e8 < 1) {
    int extra = 1 - e8;
    if (extra > 10) return (unsigned char)s;
    rshift += (unsigned)extra;
    e8 = 0;
  }
  unsigned q    = m >> rshift;
  unsigned rem  = m & ((1u << rshift) - 1u);
  unsigned halfb = 1u << (rshift - 1u);
  if (rem > halfb || (rem == halfb && (q & 1u))) q++;
  if (e8 > 0) {
    if (q >= 16u) { q >>= 1; e8++; if (e8 >= 16) return (unsigned char)(s | 0x7E); }
    return (unsigned char)(s | ((unsigned)e8 << 3) | (q & 7u));
  }
  if (q >= 8u) return (unsigned char)(s | (1u << 3));
  return (unsigned char)(s | (q & 7u));
}

__device__ __forceinline__ float sigmoidf_(float x) { return 1.0f / (1.0f + __expf(-x)); }

__global__ __launch_bounds__(TPB)
void pack_weights_fp8(const float* __restrict__ wh0,
                      const float* __restrict__ wi1,
                      const float* __restrict__ wh1,
                      unsigned char* __restrict__ ws) {
  int idx = blockIdx.x * TPB + threadIdx.x;
  if (idx < ST_WORDS)
    ((unsigned*)(ws + SOFF))[idx] = 0u;
  if (idx >= WS_SLABS) return;

  int b    = idx & 3;
  int d    = (idx >> 2)  & 7;
  int lane = (idx >> 5)  & 31;
  int kc   = (idx >> 10) & 7;
  int tile = (idx >> 13) & 7;
  int hi   = idx >> 16;
  int mat  = hi % 3;
  int wg   = hi / 3;

  int gate = tile >> 1, half = tile & 1, m = lane & 15;
  int hid  = wg * 32 + half * 16 + m;
  int grow = gate * HS2 + hid;
  int k = kc * 64 + ((d >> 2) * 32) + ((lane >> 4) * 16)
        + ((d & 3) * 4) + b;

  const float* W = (mat == 0) ? wh0 : (mat == 1 ? wh1 : wi1);
  ws[idx] = f32_to_e4m3(W[grow * HS2 + k]);
}

__device__ __forceinline__ void gbar(unsigned* cnt, unsigned* gen) {
  __syncthreads();
  __threadfence();
  if (threadIdx.x == 0) {
    unsigned sense = __hip_atomic_load(gen, __ATOMIC_RELAXED, __HIP_MEMORY_SCOPE_AGENT);
    unsigned a = __hip_atomic_fetch_add(cnt, 1u, __ATOMIC_ACQ_REL, __HIP_MEMORY_SCOPE_AGENT);
    if (a == NWG - 1) {
      __hip_atomic_store(cnt, 0u, __ATOMIC_RELAXED, __HIP_MEMORY_SCOPE_AGENT);
      __hip_atomic_store(gen, sense + 1u, __ATOMIC_RELEASE, __HIP_MEMORY_SCOPE_AGENT);
    } else {
      while (__hip_atomic_load(gen, __ATOMIC_ACQUIRE, __HIP_MEMORY_SCOPE_AGENT) == sense)
        __builtin_amdgcn_s_sleep(2);
    }
  }
  __syncthreads();
  __threadfence();
}

__device__ __forceinline__ void matvec_tiles(const char* __restrict__ slab,
                                             const unsigned* __restrict__ hv,
                                             float* __restrict__ outbuf,
                                             int wid, int lane) {
  const int tile  = wid;
  const bool act  = (lane & 15) == 0;
  const int halfk = lane >> 4;
  const int base  = act ? (halfk * 4) : 128;
  const int step  = act ? 16 : 0;
  v8f acc = {};
  for (int kc = 0; kc < 8; ++kc) {
    v8i a = *(const v8i*)(slab + (((tile * 8 + kc) * 32 + lane) << 5));
    int idx = base + kc * step;
    uint4 blo = *(const uint4*)(hv + idx);
    uint4 bhi = *(const uint4*)(hv + idx + 8);
    v8i bv;
    bv[0] = (int)blo.x; bv[1] = (int)blo.y; bv[2] = (int)blo.z; bv[3] = (int)blo.w;
    bv[4] = (int)bhi.x; bv[5] = (int)bhi.y; bv[6] = (int)bhi.z; bv[7] = (int)bhi.w;
    acc = __builtin_amdgcn_wmma_f32_16x16x64_fp8_fp8(a, bv, (short)0, acc, false, false);
  }
  if (act) {
    int gate = tile >> 1, half = tile & 1, mb = halfk * 8;
#pragma unroll
    for (int v = 0; v < 8; ++v)
      outbuf[gate * 32 + half * 16 + mb + v] = acc[v];
  }
}

__global__ __launch_bounds__(TPB)
void lstm_persistent(const float* __restrict__ event,
                     const int*   __restrict__ feat_lens,
                     const float* __restrict__ w_ih0,
                     const float* __restrict__ b_ih0,
                     const float* __restrict__ b_hh0,
                     const float* __restrict__ b_ih1,
                     const float* __restrict__ b_hh1,
                     const float* __restrict__ w_out,
                     const float* __restrict__ b_out,
                     char* __restrict__ ws,
                     float* __restrict__ out) {
  extern __shared__ char smem[];
  __shared__ float g0buf[128], p1buf[128], g1buf[128];
  __shared__ float b0s[128], b1s[128], wi0s[128];
  __shared__ unsigned h0f8w[144], h1f8w[144];
  __shared__ float h0s[32], c0s[32], h1s[32], c1s[32];
  __shared__ float lgts[16];

  const int wg = blockIdx.x, tid = threadIdx.x;
  const int lane = tid & 31, wid = tid >> 5;

  unsigned char* gh0f8 = (unsigned char*)(ws + SOFF);
  unsigned char* gh1f8 = gh0f8 + 512;
  float* gh0f32 = (float*)(ws + SOFF + 1024);
  float* gh1f32 = (float*)(ws + SOFF + 3072);
  float* evh0   = (float*)(ws + SOFF + 5120);
  float* evh1   = (float*)(ws + SOFF + 6144);
  float* evc0   = (float*)(ws + SOFF + 7168);
  float* evc1   = (float*)(ws + SOFF + 8192);
  unsigned* barc = (unsigned*)(ws + SOFF + 9216);
  unsigned* barg = barc + 1;

  {
    const float4* src = (const float4*)(ws + wg * CHUNK_BYTES);
    float4* dst = (float4*)smem;
    __builtin_prefetch(src, 0, 1);
    for (int i = tid; i < CHUNK_BYTES / 16; i += TPB) dst[i] = src[i];
  }
  if (tid < 16) { h0f8w[128 + tid] = 0u; h1f8w[128 + tid] = 0u; }
  if (tid < 128) {
    int gate = tid >> 5, r = tid & 31;
    int row = gate * HS2 + wg * 32 + r;
    b0s[tid]  = b_ih0[row] + b_hh0[row];
    b1s[tid]  = b_ih1[row] + b_hh1[row];
    wi0s[tid] = w_ih0[row];
  }
  __syncthreads();
  __builtin_amdgcn_s_cluster_barrier();
  gbar(barc, barg);

  const char* slabHH0 = smem;
  const char* slabHH1 = smem + SLAB_BYTES;
  const char* slabIH1 = smem + 2 * SLAB_BYTES;

  for (int p = 0; p < PNUM; ++p) {
    const int ln = feat_lens[p];
    if (tid < 32) {
      int j = tid, hid = wg * 32 + j;
      float nh0 = (hid < 256) ? evh0[hid] : 0.f;
      float nc0 = (hid < 256) ? evc0[hid] : 0.f;
      float nh1 = (hid < 256) ? evh1[hid] : 0.f;
      float nc1 = (hid < 256) ? evc1[hid] : 0.f;
      h0s[j] = nh0; c0s[j] = nc0; h1s[j] = nh1; c1s[j] = nc1;
      gh0f32[hid] = nh0; gh1f32[hid] = nh1;
      gh0f8[hid] = f32_to_e4m3(nh0); gh1f8[hid] = f32_to_e4m3(nh1);
    }
    gbar(barc, barg);

    for (int t = 0; t < FLEN; ++t) {
      const float x = event[p * FLEN + t];
      if (tid < 128) {
        h0f8w[tid] = ((const unsigned*)gh0f8)[tid];
        h1f8w[tid] = ((const unsigned*)gh1f8)[tid];
      }
      __syncthreads();
      matvec_tiles(slabHH0, h0f8w, g0buf, wid, lane);
      matvec_tiles(slabHH1, h1f8w, p1buf, wid, lane);
      __syncthreads();
      if (tid < 32) {
        int j = tid, hid = wg * 32 + j;
        float gi = g0buf[j]      + b0s[j]      + x * wi0s[j];
        float gf = g0buf[32 + j] + b0s[32 + j] + x * wi0s[32 + j];
        float gg = g0buf[64 + j] + b0s[64 + j] + x * wi0s[64 + j];
        float go = g0buf[96 + j] + b0s[96 + j] + x * wi0s[96 + j];
        float cn = sigmoidf_(gf) * c0s[j] + sigmoidf_(gi) * tanhf(gg);
        float hn = sigmoidf_(go) * tanhf(cn);
        if (t < ln) { c0s[j] = cn; h0s[j] = hn; }
        float hv = h0s[j];
        gh0f32[hid] = hv; gh0f8[hid] = f32_to_e4m3(hv);
      }
      gbar(barc, barg);
      if (tid < 128) h0f8w[tid] = ((const unsigned*)gh0f8)[tid];
      __syncthreads();
      matvec_tiles(slabIH1, h0f8w, g1buf, wid, lane);
      __syncthreads();
      if (tid < 32) {
        int j = tid, hid = wg * 32 + j;
        float gi = g1buf[j]      + p1buf[j]      + b1s[j];
        float gf = g1buf[32 + j] + p1buf[32 + j] + b1s[32 + j];
        float gg = g1buf[64 + j] + p1buf[64 + j] + b1s[64 + j];
        float go = g1buf[96 + j] + p1buf[96 + j] + b1s[96 + j];
        float cn = sigmoidf_(gf) * c1s[j] + sigmoidf_(gi) * tanhf(gg);
        float hn = sigmoidf_(go) * tanhf(cn);
        if (t < ln) { c1s[j] = cn; h1s[j] = hn; }
        float hv = h1s[j];
        gh1f32[hid] = hv; gh1f8[hid] = f32_to_e4m3(hv);
      }
      gbar(barc, barg);
    }
    if (tid < 32) {
      int j = tid, hid = wg * 32 + j;
      if (hid >= 256) {
        evh0[hid - 256] = h0s[j]; evc0[hid - 256] = c0s[j];
        evh1[hid - 256] = h1s[j]; evc1[hid - 256] = c1s[j];
      }
    }
    gbar(barc, barg);
  }

  if (wg == 0) {
    if (tid < OUTN) {
      float s = b_out[tid];
      const float* wr = w_out + tid * HS2;
      for (int k = 0; k < HS2; ++k) s += gh1f32[k] * wr[k];
      lgts[tid] = s;
    }
    __syncthreads();
    if (tid == 0) {
      float mx = lgts[0];
      for (int o = 1; o < OUTN; ++o) mx = fmaxf(mx, lgts[o]);
      float se = 0.f;
      for (int o = 0; o < OUTN; ++o) se += expf(lgts[o] - mx);
      float ls = logf(se);
      for (int o = 0; o < OUTN; ++o) out[o] = lgts[o] - mx - ls;
    }
  }
}

extern "C" void kernel_launch(void* const* d_in, const int* in_sizes, int n_in,
                              void* d_out, int out_size, void* d_ws, size_t ws_size,
                              hipStream_t stream) {
  const float* event = (const float*)d_in[0];
  const int*   flen  = (const int*)  d_in[1];
  const float* w_ih0 = (const float*)d_in[2];
  const float* w_hh0 = (const float*)d_in[3];
  const float* b_ih0 = (const float*)d_in[4];
  const float* b_hh0 = (const float*)d_in[5];
  const float* w_ih1 = (const float*)d_in[6];
  const float* w_hh1 = (const float*)d_in[7];
  const float* b_ih1 = (const float*)d_in[8];
  const float* b_hh1 = (const float*)d_in[9];
  const float* w_out = (const float*)d_in[10];
  const float* b_out = (const float*)d_in[11];
  (void)in_sizes; (void)n_in; (void)out_size; (void)ws_size;

  pack_weights_fp8<<<dim3(WS_SLABS / TPB), dim3(TPB), 0, stream>>>(
      w_hh0, w_ih1, w_hh1, (unsigned char*)d_ws);

  lstm_persistent<<<dim3(NWG), dim3(TPB), CHUNK_BYTES, stream>>>(
      event, flen, w_ih0, b_ih0, b_hh0, b_ih1, b_hh1, w_out, b_out,
      (char*)d_ws, (float*)d_out);
}
